// GCN_8048768712757
// MI455X (gfx1250) — matrix operand order repaired and checked by reading, not compiled, not run on hardware
//
#include <hip/hip_runtime.h>
#include <hip/hip_bf16.h>

// ---------------------------------------------------------------------------
// Relational GCN on MI455X (gfx1250, wave32).
//   emb = x @ ent_emb
//   2x: acc = sum_r A_r (emb @ W_r^T); emb = relu(acc)     (SpMM/GEMM commuted)
//   out = emb / max(||emb||_2, 1e-12)
// Dense transforms run on v_wmma_f32_16x16x32_bf16 (f32 accumulate).
// Working set (tmp + acc = 25.6 MB) stays resident in the 192 MB L2.
// ---------------------------------------------------------------------------

typedef __attribute__((ext_vector_type(16))) __bf16 v16bf;
typedef __attribute__((ext_vector_type(8)))  __bf16 v8bf;
typedef __attribute__((ext_vector_type(8)))  float  v8f;

#define NENT   50000
#define NREL   16
#define DIM    64
#define FIN    128
#define NLAYER 2
#define NEDGE  100000

static __device__ __forceinline__ unsigned short f32_to_bf16(float f) {
  unsigned int u = __float_as_uint(f);
  u += 0x7FFFu + ((u >> 16) & 1u);          // round-to-nearest-even
  return (unsigned short)(u >> 16);
}

// ---- elementwise converts ----
__global__ __launch_bounds__(256) void cvt_f32_bf16(const float* __restrict__ in,
                                                    unsigned short* __restrict__ out, int n) {
  int i = blockIdx.x * 256 + threadIdx.x;
  if (i < n) out[i] = f32_to_bf16(in[i]);
}

// out[c * rows + r] = bf16(in[r * cols + c])   (in: [rows, cols] -> out: [cols, rows])
__global__ __launch_bounds__(256) void cvt_transpose_bf16(const float* __restrict__ in,
                                                          unsigned short* __restrict__ out,
                                                          int rows, int cols) {
  int i = blockIdx.x * 256 + threadIdx.x;
  if (i < rows * cols) {
    int c = i / rows, r = i % rows;
    out[i] = f32_to_bf16(in[r * cols + c]);
  }
}

__global__ __launch_bounds__(256) void relu_cvt(const float* __restrict__ acc,
                                                float* __restrict__ emb,
                                                unsigned short* __restrict__ embh, int n) {
  int i = blockIdx.x * 256 + threadIdx.x;
  if (i < n) {
    float v = fmaxf(acc[i], 0.f);
    emb[i] = v;
    embh[i] = f32_to_bf16(v);
  }
}

// ---- WMMA GEMM: C[M,64] = A[M,K] (bf16 row-major) @ Bt[64,K]^T (bf16 row-major) ----
// grid.x = M/16 row tiles; 4 waves per block, wave w owns 16-column tile w.
// Per-lane fragments use contiguous K order (k = 16*hi + s) for A(16x32)/B(32x16):
//   A: lane<16 row m holds K {0..15},       lane>=16 holds K {16..31}
//   B: lane<16 col n holds K {0..15},       lane>=16 holds K {16..31}
// Both become contiguous loads because B is stored transposed.
__global__ __launch_bounds__(128) void gemm_bf16_wmma(const __bf16* __restrict__ A,
                                                      const __bf16* __restrict__ Bt,
                                                      float* __restrict__ C, int K) {
  const int lane  = threadIdx.x & 31;
  const int wtile = threadIdx.x >> 5;        // column tile 0..3
  const int m     = lane & 15;
  const int hi    = lane >> 4;               // lane half
  const int row   = blockIdx.x * 16 + m;
  const int ncol  = wtile * 16 + m;

  const __bf16* arow = A  + (size_t)row  * K + (hi ? 16 : 0);
  const __bf16* brow = Bt + (size_t)ncol * K + (hi ? 16 : 0);

  v8f c = {};
  for (int kb = 0; kb < K; kb += 32) {
    v8bf  alo = *(const v8bf*)(arow + kb);          // K offsets +0..7
    v8bf  ahi = *(const v8bf*)(arow + kb + 8);      // K offsets +8..15
    v16bf a   = __builtin_shufflevector(alo, ahi, 0,1,2,3,4,5,6,7,8,9,10,11,12,13,14,15);
    v16bf b   = *(const v16bf*)(brow + kb);         // 16 contiguous K values
    c = __builtin_amdgcn_wmma_f32_16x16x32_bf16(false, a, false, b, (short)0, c,
                                                false, false);
  }

  // 32-bit C/D layout: VGPR v -> (M = v + 8*hi, N = lane%16)
  const int mbase = blockIdx.x * 16 + hi * 8;
  float* cptr = C + (size_t)mbase * DIM + wtile * 16 + m;
#pragma unroll
  for (int v = 0; v < 8; ++v) cptr[(size_t)v * DIM] = c[v];
}

// ---- SpMM scatter: acc[row] += val * T[col]  (one wave32 per edge) ----
__global__ __launch_bounds__(256) void spmm_scatter(const float* __restrict__ T,
                                                    const int*   __restrict__ rows,
                                                    const int*   __restrict__ cols,
                                                    const float* __restrict__ vals,
                                                    float* __restrict__ acc, int nEdges) {
  const int lane = threadIdx.x & 31;
  const int e    = blockIdx.x * 8 + (threadIdx.x >> 5);
  if (e >= nEdges) return;
  const int   row = rows[e];                 // uniform per wave -> scalar loads
  const int   col = cols[e];
  const float val = vals[e];
  float2 v = ((const float2*)(T + (size_t)col * DIM))[lane];
  float* dst = acc + (size_t)row * DIM + lane * 2;
  unsafeAtomicAdd(dst,     val * v.x);       // global_atomic_add_f32, L2-resident
  unsafeAtomicAdd(dst + 1, val * v.y);
}

// ---- row-wise L2 normalize (one wave32 per row) ----
__global__ __launch_bounds__(256) void l2_normalize(const float* __restrict__ emb,
                                                    float* __restrict__ out, int n) {
  const int lane = threadIdx.x & 31;
  const int row  = blockIdx.x * 8 + (threadIdx.x >> 5);
  if (row >= n) return;
  float2 v = ((const float2*)(emb + (size_t)row * DIM))[lane];
  float s = v.x * v.x + v.y * v.y;
#pragma unroll
  for (int off = 16; off; off >>= 1) s += __shfl_xor(s, off, 32);
  float inv = 1.f / fmaxf(sqrtf(s), 1e-12f);
  ((float2*)(out + (size_t)row * DIM))[lane] = make_float2(v.x * inv, v.y * inv);
}

extern "C" void kernel_launch(void* const* d_in, const int* in_sizes, int n_in,
                              void* d_out, int out_size, void* d_ws, size_t ws_size,
                              hipStream_t stream) {
  const float* x         = (const float*)d_in[0];   // [N, FIN]
  const float* ent_emb   = (const float*)d_in[1];   // [FIN, D]
  const float* rel_trans = (const float*)d_in[2];   // [L, R, D, D]
  const int*   edge_row  = (const int*)  d_in[3];   // [R, E]
  const int*   edge_col  = (const int*)  d_in[4];   // [R, E]
  const float* edge_val  = (const float*)d_in[5];   // [R, E]
  float* out = (float*)d_out;

  char* ws = (char*)d_ws;
  size_t off = 0;
  auto alloc = [&](size_t bytes) -> void* {
    void* p = ws + off;
    off = (off + bytes + 255) & ~(size_t)255;
    return p;
  };
  unsigned short* xh   = (unsigned short*)alloc((size_t)NENT * FIN * 2);          // bf16(x)
  unsigned short* b0h  = (unsigned short*)alloc((size_t)DIM * FIN * 2);           // bf16(ent_emb^T)
  unsigned short* wh   = (unsigned short*)alloc((size_t)NLAYER * NREL * DIM * DIM * 2);
  unsigned short* embh = (unsigned short*)alloc((size_t)NENT * DIM * 2);          // bf16(emb)
  float* emb = (float*)alloc((size_t)NENT * DIM * 4);
  float* tmp = (float*)alloc((size_t)NENT * DIM * 4);
  float* acc = (float*)alloc((size_t)NENT * DIM * 4);
  (void)in_sizes; (void)n_in; (void)out_size; (void)ws_size;

  // one-time precision conversion of operands
  cvt_f32_bf16     <<<(NENT * FIN + 255) / 256, 256, 0, stream>>>(x, xh, NENT * FIN);
  cvt_transpose_bf16<<<(FIN * DIM + 255) / 256, 256, 0, stream>>>(ent_emb, b0h, FIN, DIM);
  cvt_f32_bf16     <<<(NLAYER * NREL * DIM * DIM + 255) / 256, 256, 0, stream>>>(
      rel_trans, wh, NLAYER * NREL * DIM * DIM);

  // emb = x @ ent_emb   (K = 128; N = 50000 = 3125 * 16 exactly)
  gemm_bf16_wmma<<<NENT / 16, 128, 0, stream>>>((const __bf16*)xh, (const __bf16*)b0h,
                                                emb, FIN);
  cvt_f32_bf16<<<(NENT * DIM + 255) / 256, 256, 0, stream>>>(emb, embh, NENT * DIM);

  for (int l = 0; l < NLAYER; ++l) {
    hipMemsetAsync(acc, 0, (size_t)NENT * DIM * 4, stream);
    for (int r = 0; r < NREL; ++r) {
      // transform first: tmp = emb @ W_r^T  (Bt == W_r row-major, no transpose needed)
      const __bf16* W = (const __bf16*)(wh + (size_t)(l * NREL + r) * DIM * DIM);
      gemm_bf16_wmma<<<NENT / 16, 128, 0, stream>>>((const __bf16*)embh, W, tmp, DIM);
      // then scatter: acc[row] += val * tmp[col]   (tmp+acc stay in L2)
      spmm_scatter<<<(NEDGE + 7) / 8, 256, 0, stream>>>(
          tmp, edge_row + r * NEDGE, edge_col + r * NEDGE, edge_val + r * NEDGE,
          acc, NEDGE);
    }
    relu_cvt<<<(NENT * DIM + 255) / 256, 256, 0, stream>>>(acc, emb, embh, NENT * DIM);
  }

  l2_normalize<<<(NENT + 7) / 8, 256, 0, stream>>>(emb, out, NENT);
}
